// NetVladCNN_13572096655815
// MI455X (gfx1250) — matrix operand order repaired and checked by reading, not compiled, not run on hardware
//
#include <hip/hip_runtime.h>
#include <hip/hip_bf16.h>
#include <math.h>

// Problem constants (B,D,H,W,K) = (64,512,32,32,64), N = H*W = 1024
#define Bsz 64
#define Dd  512
#define Hh  32
#define Nn  1024
#define Kk  64

typedef __attribute__((ext_vector_type(16))) __bf16 v16bf;
typedef __attribute__((ext_vector_type(8)))  float  v8f;

union AFrag { v16bf v; unsigned short u[16]; };
union CFrag { v8f v; float f[8]; };

__device__ __forceinline__ unsigned short f2bf(float f) {
  return __builtin_bit_cast(unsigned short, static_cast<__bf16>(f));
}
__device__ __forceinline__ unsigned pk2bf(float lo, float hi) {
  unsigned a = f2bf(lo), b = f2bf(hi);
  return a | (b << 16);
}

// CDNA5 async global->LDS copy (ASYNCcnt path), with safe fallback.
#if defined(__has_builtin)
#if __has_builtin(__builtin_amdgcn_global_load_async_to_lds_b128) && \
    __has_builtin(__builtin_amdgcn_s_wait_asynccnt)
#define HAVE_ASYNC_LDS 1
#endif
#endif
#ifndef HAVE_ASYNC_LDS
#define HAVE_ASYNC_LDS 0
#endif

#if HAVE_ASYNC_LDS
typedef __attribute__((ext_vector_type(4))) int vi4;
typedef __attribute__((address_space(1))) vi4 gvi4;   // global int4
typedef __attribute__((address_space(3))) vi4 lvi4;   // LDS int4
#endif

// ---------------------------------------------------------------------------
// Kernel 1: feat = W(64x512) @ x_b(512x1024) tile, softmax over H (block-local
// because each block owns 4 W-columns x all 32 H rows), store a as bf16.
// b_conv is omitted: softmax over H is invariant to a per-k constant.
// ---------------------------------------------------------------------------
__global__ __launch_bounds__(256)
void k1_conv_softmax(const float* __restrict__ x, const float* __restrict__ w,
                     unsigned short* __restrict__ a_bf) {
  __shared__ unsigned short lds_w [Kk * 32];    //  4 KB [k][dd]
  __shared__ unsigned short lds_xt[128 * 32];   //  8 KB [p][dd]  (transposed!)
  __shared__ float          lds_f [Kk * 128];   // 32 KB [k][p], p = h*4 + wc
  __shared__ unsigned short lds_an[Kk * 128];   // 16 KB [k][p] softmax as bf16

  const int b    = blockIdx.y;
  const int wc0  = blockIdx.x * 4;              // 4 W-columns per block
  const int tid  = threadIdx.x;
  const int lane = tid & 31;
  const int wave = tid >> 5;
  const int kt   = wave >> 1;                   // K tile (0..3)
  const int nh   = wave & 1;                    // N half (0..1)
  const int m    = lane & 15;
  const int halfL= lane >> 4;

  CFrag acc[4];
  #pragma unroll
  for (int i = 0; i < 4; ++i)
    #pragma unroll
    for (int j = 0; j < 8; ++j) acc[i].f[j] = 0.f;

  const int kw  = tid >> 2;                     // 0..63 (w-row for staging)
  const int seg = tid & 3;                      // 0..3
  const int qx  = tid & 31;                     // d-row for x staging
  const int hg  = tid >> 5;                     // h-group (0..7)

  for (int ds = 0; ds < 16; ++ds) {
    const int d0 = ds * 32;
    __syncthreads();
    { // stage W tile 64x32 -> bf16 (row-major [k][dd])
      const float4* wp = (const float4*)(w + kw * Dd + d0 + seg * 8);
      float4 w0 = wp[0], w1 = wp[1];
      unsigned* dst = (unsigned*)&lds_w[kw * 32 + seg * 8];
      dst[0] = pk2bf(w0.x, w0.y); dst[1] = pk2bf(w0.z, w0.w);
      dst[2] = pk2bf(w1.x, w1.y); dst[3] = pk2bf(w1.z, w1.w);
    }
    { // stage x tile TRANSPOSED: lds_xt[p][dd], p = h*4 + wl
      #pragma unroll
      for (int i = 0; i < 4; ++i) {
        const int h = hg * 4 + i;
        float4 v = *(const float4*)(x + ((size_t)b * Dd + d0 + qx) * Nn + h * 32 + wc0);
        lds_xt[(h * 4 + 0) * 32 + qx] = f2bf(v.x);
        lds_xt[(h * 4 + 1) * 32 + qx] = f2bf(v.y);
        lds_xt[(h * 4 + 2) * 32 + qx] = f2bf(v.z);
        lds_xt[(h * 4 + 3) * 32 + qx] = f2bf(v.w);
      }
      if (ds < 15)  // WGP-scope prefetch of next D-step's x row (L2-resident)
        __builtin_prefetch(x + ((size_t)b * Dd + d0 + 32 + qx) * Nn + hg * 128, 0, 3);
    }
    __syncthreads();
    AFrag av;                                   // A = W rows (16x32)
    #pragma unroll
    for (int e = 0; e < 16; ++e) {
      int kk = e + 16 * halfL;
      av.u[e] = lds_w[(kt * 16 + m) * 32 + kk];
    }
    #pragma unroll
    for (int bt = 0; bt < 4; ++bt) {            // B = x cols (32x16)
      AFrag bv;
      const int col = nh * 64 + bt * 16 + m;
      #pragma unroll
      for (int e = 0; e < 16; ++e)              // contiguous 32B run per lane
        bv.u[e] = lds_xt[col * 32 + 16 * halfL + e];
      acc[bt].v = __builtin_amdgcn_wmma_f32_16x16x32_bf16(
          false, av.v, false, bv.v, (short)0, acc[bt].v, false, false);
    }
  }
  __syncthreads();
  #pragma unroll
  for (int bt = 0; bt < 4; ++bt) {              // spill feat tile to LDS
    const int p = nh * 64 + bt * 16 + m;
    #pragma unroll
    for (int v = 0; v < 8; ++v)
      lds_f[(kt * 16 + v + 8 * halfL) * 128 + p] = acc[bt].f[v];
  }
  __syncthreads();
  { // softmax over H for column (k = tid/4, wc = tid%4); result -> lds_an
    const int k  = tid >> 2;
    const int wc = tid & 3;
    float mx = -3.4e38f;
    for (int h = 0; h < 32; ++h) mx = fmaxf(mx, lds_f[k * 128 + h * 4 + wc]);
    float s = 0.f;
    for (int h = 0; h < 32; ++h) s += __expf(lds_f[k * 128 + h * 4 + wc] - mx);
    const float rs = 1.0f / s;
    for (int h = 0; h < 32; ++h)
      lds_an[k * 128 + h * 4 + wc] = f2bf(__expf(lds_f[k * 128 + h * 4 + wc] - mx) * rs);
  }
  __syncthreads();
  #pragma unroll
  for (int i = 0; i < 8; ++i) {                 // wide 8B stores of a (bf16)
    const int idx = i * 256 + tid;              // 0..2047
    const int k   = idx >> 5;
    const int h   = idx & 31;
    *(uint2*)(a_bf + ((size_t)(b * Kk + k)) * Nn + h * 32 + wc0) =
        *(const uint2*)&lds_an[k * 128 + h * 4];
  }
}

// ---------------------------------------------------------------------------
// Kernel 2: V_b(64x512) = a_b(64x1024) @ x_b^T(1024x512) - 32*c ; fused
// L2-normalize over K (all 64 k live in the block) ; write out[(d,k,b)].
// a.sum(-1) == W == 32 exactly (softmax over H sums to 1 per W-column).
// a-tile staging uses CDNA5 GLOBAL_LOAD_ASYNC_TO_LDS_B128 when available.
// ---------------------------------------------------------------------------
__global__ __launch_bounds__(256)
void k2_vlad_norm(const float* __restrict__ x, const float* __restrict__ c,
                  const unsigned short* __restrict__ a_bf, float* __restrict__ out) {
  __shared__ unsigned short lds_a[Kk * 32];     //  4 KB [k][nn]
  __shared__ unsigned short lds_x[128 * 32];    //  8 KB [dd][nn]
  __shared__ float          lds_v[Kk * 128];    // 32 KB [k][dd]
  __shared__ float          lds_rn[128];

  const int b    = blockIdx.y;
  const int d0   = blockIdx.x * 128;
  const int tid  = threadIdx.x;
  const int lane = tid & 31;
  const int wave = tid >> 5;
  const int kt   = wave >> 1;                   // K tile (0..3)
  const int dh   = wave & 1;                    // D half (0..1)
  const int m    = lane & 15;
  const int halfL= lane >> 4;

  CFrag acc[4];
  #pragma unroll
  for (int i = 0; i < 4; ++i)
    #pragma unroll
    for (int j = 0; j < 8; ++j) acc[i].f[j] = 0.f;

  const int ka  = tid >> 2;                     // a staging row
  const int seg = tid & 3;
  const int dd  = tid >> 1;                     // x staging row
  const int hf  = tid & 1;

  for (int ns = 0; ns < 32; ++ns) {
    const int n0 = ns * 32;
    __syncthreads();
    { // stage a tile 64x32 bf16: pure copy -> async DMA to LDS (ASYNCcnt)
      const unsigned short* src = a_bf + ((size_t)(b * Kk + ka)) * Nn + n0 + seg * 8;
      unsigned short* dst = &lds_a[ka * 32 + seg * 8];
#if HAVE_ASYNC_LDS
      __builtin_amdgcn_global_load_async_to_lds_b128(
          (gvi4*)src, (lvi4*)dst, 0, 0);
#else
      *(uint4*)dst = *(const uint4*)src;
#endif
    }
    { // stage x tile 128(d) x 32(n) -> bf16, packed b128 LDS stores
      const float4* xp = (const float4*)(x + ((size_t)b * Dd + d0 + dd) * Nn + n0 + hf * 16);
      float4 v0 = xp[0], v1 = xp[1], v2 = xp[2], v3 = xp[3];
      uint4 p0, p1;
      p0.x = pk2bf(v0.x, v0.y); p0.y = pk2bf(v0.z, v0.w);
      p0.z = pk2bf(v1.x, v1.y); p0.w = pk2bf(v1.z, v1.w);
      p1.x = pk2bf(v2.x, v2.y); p1.y = pk2bf(v2.z, v2.w);
      p1.z = pk2bf(v3.x, v3.y); p1.w = pk2bf(v3.z, v3.w);
      uint4* dst = (uint4*)&lds_x[dd * 32 + hf * 16];
      dst[0] = p0; dst[1] = p1;
      if (ns < 31)  // WGP-scope prefetch of next n-tile of this d row
        __builtin_prefetch(x + ((size_t)b * Dd + d0 + dd) * Nn + n0 + 32 + hf * 16, 0, 3);
    }
#if HAVE_ASYNC_LDS
    __builtin_amdgcn_s_wait_asynccnt(0);        // barrier does not cover ASYNCcnt
#endif
    __syncthreads();
    AFrag av;                                   // A = a rows (16x32)
    #pragma unroll
    for (int e = 0; e < 16; ++e) {
      int kk = e + 16 * halfL;
      av.u[e] = lds_a[(kt * 16 + m) * 32 + kk];
    }
    #pragma unroll
    for (int bt = 0; bt < 4; ++bt) {            // B = x^T cols (32x16)
      AFrag bv;
      const int dc = dh * 64 + bt * 16 + m;
      #pragma unroll
      for (int e = 0; e < 16; ++e)              // contiguous 32B run per lane
        bv.u[e] = lds_x[dc * 32 + 16 * halfL + e];
      acc[bt].v = __builtin_amdgcn_wmma_f32_16x16x32_bf16(
          false, av.v, false, bv.v, (short)0, acc[bt].v, false, false);
    }
  }
  __syncthreads();
  #pragma unroll
  for (int bt = 0; bt < 4; ++bt) {              // V = acc - 32*c -> LDS
    const int dc = dh * 64 + bt * 16 + m;
    #pragma unroll
    for (int v = 0; v < 8; ++v) {
      const int krow = kt * 16 + v + 8 * halfL;
      lds_v[krow * 128 + dc] = acc[bt].f[v] - 32.0f * c[krow * Dd + d0 + dc];
    }
  }
  __syncthreads();
  if (tid < 128) {                              // L2 norm over K per d-column
    float s = 0.f;
    for (int k = 0; k < Kk; ++k) { float v = lds_v[k * 128 + tid]; s += v * v; }
    lds_rn[tid] = 1.0f / fmaxf(sqrtf(s), 1e-12f);
  }
  __syncthreads();
  #pragma unroll
  for (int i = 0; i < 32; ++i) {                // out[(d,k,b)] = V * rnorm
    const int idx = i * 256 + tid;
    const int k   = idx & 63;
    const int ddx = idx >> 6;
    out[(size_t)(d0 + ddx) * (Kk * Bsz) + k * Bsz + b] = lds_v[k * 128 + ddx] * lds_rn[ddx];
  }
}

extern "C" void kernel_launch(void* const* d_in, const int* in_sizes, int n_in,
                              void* d_out, int out_size, void* d_ws, size_t ws_size,
                              hipStream_t stream) {
  (void)in_sizes; (void)n_in; (void)out_size; (void)ws_size;
  const float* x = (const float*)d_in[0];
  const float* w = (const float*)d_in[1];
  // d_in[2] = b_conv: constant along the softmax (H) axis -> cancels, unused.
  const float* c = (const float*)d_in[3];
  float* out = (float*)d_out;
  unsigned short* a_bf = (unsigned short*)d_ws;  // B*K*N bf16 = 8.4 MB scratch

  k1_conv_softmax<<<dim3(8, Bsz), 256, 0, stream>>>(x, w, a_bf);
  k2_vlad_norm  <<<dim3(4, Bsz), 256, 0, stream>>>(x, c, a_bf, out);
}
